// MllamaTextSelfAttention_33088428049115
// MI455X (gfx1250) — matrix operand order repaired and checked by reading, not compiled, not run on hardware
//
#include <hip/hip_runtime.h>

// Problem constants (fixed by the reference setup)
#define B_DIM 2
#define S_DIM 2048
#define H_DIM 4096
#define NH    32
#define NKV   8
#define HD    128
#define M_TOT (B_DIM * S_DIM)   // 4096 rows for all GEMMs

typedef __bf16 bf16_t;
typedef bf16_t v16bf __attribute__((ext_vector_type(16)));
typedef float  v8f   __attribute__((ext_vector_type(8)));
typedef unsigned int u32x4 __attribute__((ext_vector_type(4)));
typedef int          i32x8 __attribute__((ext_vector_type(8)));
typedef int          i32x4 __attribute__((ext_vector_type(4)));

union FragBF { v16bf v; uint4 q[2]; };

#if defined(__gfx1250__) && __has_builtin(__builtin_amdgcn_tensor_load_to_lds) && \
    __has_builtin(__builtin_amdgcn_s_wait_tensorcnt)
#define USE_TDM 1
#else
#define USE_TDM 0
#endif

__device__ __forceinline__ unsigned short f2bf(float f) {
  unsigned int u = __float_as_uint(f);
  u += 0x7FFFu + ((u >> 16) & 1u);  // round-to-nearest-even
  return (unsigned short)(u >> 16);
}
__device__ __forceinline__ float bf2f(unsigned short b) {
  return __uint_as_float(((unsigned int)b) << 16);
}

#if USE_TDM
// ---------------------------------------------------------------------------
// TDM 2D tile load: global (bf16, row stride stride0 elements) -> LDS, packed.
// D# built per CDNA5 ISA (group0 128b, group1 256b).
// Tensor dims set to 2^30 (tiles never clipped; all shapes divide evenly).
// Toolchain exposes the 6-arg builtin: (g0, g1, g2, g3, g4, cpol).
// ---------------------------------------------------------------------------
__device__ __forceinline__ void tdm_load_2d(void* lds_ptr, const void* gptr,
                                            unsigned int tile_d0,
                                            unsigned int tile_d1,
                                            unsigned long long stride0) {
  const unsigned long long ga = (unsigned long long)(size_t)gptr;
  const unsigned int lds_off = (unsigned int)(size_t)lds_ptr;  // LDS aperture: low 32b
  u32x4 g0;
  g0[0] = 1u;                                                    // count=1 (valid)
  g0[1] = lds_off;                                               // lds_addr
  g0[2] = (unsigned int)ga;                                      // global_addr[31:0]
  g0[3] = (unsigned int)((ga >> 32) & 0x01FFFFFFu) | (2u << 30); // addr[56:32]|type=2
  i32x8 g1;
  g1[0] = (int)(1u << 16);                  // wg_mask=0, data_size=1 (2 bytes)
  g1[1] = 0;                                // barrier_addr=0, tensor_dim0[15:0]=0
  g1[2] = (int)0x4000;                      // tensor_dim0=2^30, tensor_dim1[15:0]=0
  g1[3] = (int)(0x4000u | (tile_d0 << 16)); // tensor_dim1=2^30, tile_dim0
  g1[4] = (int)tile_d1;                     // tile_dim1; tile_dim2=0
  g1[5] = (int)(unsigned int)stride0;       // tensor_dim0_stride[31:0]
  g1[6] = (int)((stride0 >> 32) & 0xFFFFu); // stride[47:32]; dim1_stride[15:0]=0
  g1[7] = 0;
  i32x4 z4 = {0, 0, 0, 0};
  i32x8 z8 = {0, 0, 0, 0, 0, 0, 0, 0};
  __builtin_amdgcn_tensor_load_to_lds(g0, g1, z4, z4, z8, 0);
}
#endif

// ---------------------------------------------------------------------------
// One-time converters (hoisted out of GEMM hot loops):
//   cvt_f32_bf16      : fp32 [R][C] -> bf16 same layout (vectorized)
//   cvt_w_transpose   : fp32 W [N][K] -> bf16 WT [K][N]
// ---------------------------------------------------------------------------
__global__ __launch_bounds__(256) void cvt_f32_bf16(
    const float* __restrict__ src, unsigned short* __restrict__ dst) {
  const size_t i = ((size_t)blockIdx.x * 256 + threadIdx.x) * 8;
  float4 a = *(const float4*)(src + i);
  float4 b = *(const float4*)(src + i + 4);
  uint4 o;
  o.x = (unsigned int)f2bf(a.x) | ((unsigned int)f2bf(a.y) << 16);
  o.y = (unsigned int)f2bf(a.z) | ((unsigned int)f2bf(a.w) << 16);
  o.z = (unsigned int)f2bf(b.x) | ((unsigned int)f2bf(b.y) << 16);
  o.w = (unsigned int)f2bf(b.z) | ((unsigned int)f2bf(b.w) << 16);
  *(uint4*)(dst + i) = o;
}

__global__ __launch_bounds__(256) void cvt_w_transpose(
    const float* __restrict__ src, unsigned short* __restrict__ dst, int N,
    int K) {
  const size_t idx = (size_t)blockIdx.x * 256 + threadIdx.x;  // one per (n,k4)
  const int kq = (int)(idx % (size_t)(K / 4)) * 4;
  const int n  = (int)(idx / (size_t)(K / 4));
  float4 f = *(const float4*)(src + (size_t)n * K + kq);
  dst[(size_t)(kq + 0) * N + n] = f2bf(f.x);
  dst[(size_t)(kq + 1) * N + n] = f2bf(f.y);
  dst[(size_t)(kq + 2) * N + n] = f2bf(f.z);
  dst[(size_t)(kq + 3) * N + n] = f2bf(f.w);
}

// ---------------------------------------------------------------------------
// Tiled GEMM:  C[M,N] = A[M,K] * BT[K,N]    (A bf16 row-major, BT bf16 [K][N])
// 256 threads = 8 waves; tile 128x128, K-step 32; wave = 64x32 (4x2 frags).
// LDS double-buffered; tiles staged by the Tensor Data Mover (wave 0 issues,
// s_wait_tensorcnt + barrier), or vector-copied when TDM builtin unavailable.
// ---------------------------------------------------------------------------
template <bool C_STORE_BF16>
__global__ __launch_bounds__(256) void gemm_bf16_wmma(
    const unsigned short* __restrict__ A, const unsigned short* __restrict__ BT,
    void* __restrict__ Cv, int M, int N, int K) {
  __shared__ unsigned short ldsA[2][128 * 32];  // [m][k]
  __shared__ unsigned short ldsB[2][32 * 128];  // [k][n]

  const int tid   = threadIdx.x;
  const int lane  = tid & 31;
  const int wave  = tid >> 5;
  const int m0    = blockIdx.y * 128;
  const int n0    = blockIdx.x * 128;
  const int waveM = (wave >> 2) * 64;
  const int waveN = (wave & 3) * 32;
  const int lm    = lane & 15;
  const int hi    = lane >> 4;
  const int h8    = hi * 8;

  v8f z = {};
  v8f c[4][2];
#pragma unroll
  for (int i = 0; i < 4; ++i)
#pragma unroll
    for (int j = 0; j < 2; ++j) c[i][j] = z;

  const int nk = K / 32;
#if USE_TDM
  if (wave == 0) {
    tdm_load_2d(&ldsA[0][0], A + (size_t)m0 * K, 32, 128, (unsigned long long)K);
    tdm_load_2d(&ldsB[0][0], BT + n0, 128, 32, (unsigned long long)N);
  }
#endif

  for (int it = 0; it < nk; ++it) {
    const int cur = it & 1;
#if USE_TDM
    if (wave == 0) __builtin_amdgcn_s_wait_tensorcnt((short)0);
    __syncthreads();  // current buffer DMA visible; prev buffer fully consumed
    if (wave == 0 && it + 1 < nk) {
      const int k1 = (it + 1) * 32;
      tdm_load_2d(&ldsA[cur ^ 1][0], A + (size_t)m0 * K + k1, 32, 128,
                  (unsigned long long)K);
      tdm_load_2d(&ldsB[cur ^ 1][0], BT + (size_t)k1 * N + n0, 128, 32,
                  (unsigned long long)N);
    }
#else
    {
      const int k0 = it * 32;
      const int row = tid >> 1, cseg = (tid & 1) * 16;
      const unsigned short* sa = A + (size_t)(m0 + row) * K + k0 + cseg;
      *(uint4*)&ldsA[cur][row * 32 + cseg]     = *(const uint4*)(sa);
      *(uint4*)&ldsA[cur][row * 32 + cseg + 8] = *(const uint4*)(sa + 8);
      const int kr = tid >> 3, nseg = (tid & 7) * 16;
      const unsigned short* sb = BT + (size_t)(k0 + kr) * N + n0 + nseg;
      *(uint4*)&ldsB[cur][kr * 128 + nseg]     = *(const uint4*)(sb);
      *(uint4*)&ldsB[cur][kr * 128 + nseg + 8] = *(const uint4*)(sb + 8);
    }
    __syncthreads();
#endif

    FragBF a[4], b[2];
#pragma unroll
    for (int i = 0; i < 4; ++i) {
      const unsigned short* p = &ldsA[cur][(waveM + 16 * i + lm) * 32 + h8];
      a[i].q[0] = *(const uint4*)(p);
      a[i].q[1] = *(const uint4*)(p + 16);
    }
#pragma unroll
    for (int j = 0; j < 2; ++j) {
      const unsigned short* p = &ldsB[cur][waveN + 16 * j + lm];
#pragma unroll
      for (int s = 0; s < 16; ++s)
        ((unsigned short*)&b[j])[s] = p[(h8 + (s & 7) + ((s >> 3) << 4)) * 128];
    }
#pragma unroll
    for (int i = 0; i < 4; ++i)
#pragma unroll
      for (int j = 0; j < 2; ++j)
        c[i][j] = __builtin_amdgcn_wmma_f32_16x16x32_bf16(
            false, a[i].v, false, b[j].v, (short)0, c[i][j], false, false);
    __syncthreads();  // consumers done before this buffer is re-filled
  }

  // epilogue: C/D layout m = r + 8*(lane>>4), n = lane&15
#pragma unroll
  for (int i = 0; i < 4; ++i)
#pragma unroll
    for (int j = 0; j < 2; ++j)
#pragma unroll
      for (int r = 0; r < 8; ++r) {
        const size_t row = (size_t)(m0 + waveM + 16 * i + r + 8 * hi);
        const size_t col = (size_t)(n0 + waveN + 16 * j + lm);
        if constexpr (C_STORE_BF16)
          ((unsigned short*)Cv)[row * N + col] = f2bf(c[i][j][r]);
        else
          ((float*)Cv)[row * N + col] = c[i][j][r];
      }
}

// ---------------------------------------------------------------------------
// RoPE + [B,S,h,d] -> [B,h,S,d] transpose (bf16 in / bf16 out).
// ---------------------------------------------------------------------------
__global__ __launch_bounds__(256) void rope_transpose(
    const unsigned short* __restrict__ lin, const float* __restrict__ cosb,
    const float* __restrict__ sinb, unsigned short* __restrict__ outp, int nh,
    int applyRope) {
  const int idx = blockIdx.x * blockDim.x + threadIdx.x;
  const int dp  = idx & 63;
  int rest      = idx >> 6;
  const int s   = rest & (S_DIM - 1);
  rest >>= 11;  // S_DIM = 2048
  const int h = rest % nh;
  const int b = rest / nh;

  const size_t ibase = ((size_t)(b * S_DIM + s) * nh + h) * HD;
  const float x1 = bf2f(lin[ibase + dp]);
  const float x2 = bf2f(lin[ibase + dp + 64]);
  float y1 = x1, y2 = x2;
  if (applyRope) {
    const float c1 = cosb[s * HD + dp];
    const float s1 = sinb[s * HD + dp];
    const float c2 = cosb[s * HD + dp + 64];
    const float s2 = sinb[s * HD + dp + 64];
    y1 = x1 * c1 - x2 * s1;
    y2 = x2 * c2 + x1 * s2;
  }
  const size_t obase = ((size_t)(b * nh + h) * S_DIM + s) * HD;
  outp[obase + dp]      = f2bf(y1);
  outp[obase + dp + 64] = f2bf(y2);
}

// ---------------------------------------------------------------------------
// Flash attention (causal, GQA group=4), bf16 WMMA, fp32 online softmax.
// Block = 128 threads = 4 waves; block owns 64 q rows, wave owns 16.
// V tile (contiguous 8KB) staged by TDM; K tile transposed manually.
// ---------------------------------------------------------------------------
__global__ __launch_bounds__(128) void flash_attn(
    const unsigned short* __restrict__ Qm, const unsigned short* __restrict__ Km,
    const unsigned short* __restrict__ Vm, unsigned short* __restrict__ Om) {
  __shared__ unsigned short ldsK[HD * 32];     // [d][kc]
  __shared__ unsigned short ldsV[32 * HD];     // [kc][d]
  __shared__ unsigned short ldsP[4][16 * 32];  // per-wave P tile

  const int tid  = threadIdx.x;
  const int lane = tid & 31;
  const int wave = tid >> 5;
  const int lm   = lane & 15;
  const int hi   = lane >> 4;
  const int h8   = hi * 8;

  const int bh  = blockIdx.y;
  const int b   = bh >> 5;
  const int h   = bh & 31;
  const int hkv = h >> 2;  // GQA: jnp.repeat(k, 4, axis=1)
  const int qt  = blockIdx.x;
  const int q0  = qt * 64 + wave * 16;

  FragBF qf[4];
  {
    const unsigned short* qb =
        Qm + ((size_t)(b * NH + h) * S_DIM + q0 + lm) * HD;
#pragma unroll
    for (int dc = 0; dc < 4; ++dc) {
      qf[dc].q[0] = *(const uint4*)(qb + dc * 32 + h8);
      qf[dc].q[1] = *(const uint4*)(qb + dc * 32 + h8 + 16);
    }
  }

  v8f z = {};
  v8f o[8];
#pragma unroll
  for (int j = 0; j < 8; ++j) o[j] = z;
  float mrow[8], lrow[8];
#pragma unroll
  for (int r = 0; r < 8; ++r) { mrow[r] = -3.0e38f; lrow[r] = 0.0f; }

  const float scale = 0.08838834764831845f;  // 1/sqrt(128)
  const int nblk = qt * 2 + 2;

  for (int it = 0; it < nblk; ++it) {
    const int kc0 = it * 32;
    const unsigned short* kvbase =
        Km + ((size_t)(b * NKV + hkv) * S_DIM + kc0) * HD;
    const unsigned short* vvbase =
        Vm + ((size_t)(b * NKV + hkv) * S_DIM + kc0) * HD;
#if USE_TDM
    if (wave == 0) tdm_load_2d(&ldsV[0], vvbase, 4096, 1, 4096ull);
#endif
    {
      const int kc = tid >> 2;
      const int d0 = (tid & 3) * 32;
      const unsigned short* ksrc = kvbase + (size_t)kc * HD + d0;
#pragma unroll
      for (int i = 0; i < 32; i += 8) {
        uint4 x = *(const uint4*)(ksrc + i);
        const unsigned short* xs = (const unsigned short*)&x;
#pragma unroll
        for (int e = 0; e < 8; ++e) ldsK[(d0 + i + e) * 32 + kc] = xs[e];
      }
#if !USE_TDM
      const unsigned short* vsrc = vvbase + (size_t)kc * HD + d0;
#pragma unroll
      for (int i = 0; i < 32; i += 8)
        *(uint4*)&ldsV[kc * HD + d0 + i] = *(const uint4*)(vsrc + i);
#endif
    }
#if USE_TDM
    if (wave == 0) __builtin_amdgcn_s_wait_tensorcnt((short)0);
#endif
    __syncthreads();

    // scores S = Q K^T (two 16-col tiles)
    v8f sfrag[2];
    sfrag[0] = z; sfrag[1] = z;
#pragma unroll
    for (int dc = 0; dc < 4; ++dc) {
#pragma unroll
      for (int j = 0; j < 2; ++j) {
        FragBF kb;
        const unsigned short* p = &ldsK[(dc * 32) * 32 + 16 * j + lm];
#pragma unroll
        for (int s = 0; s < 16; ++s)
          ((unsigned short*)&kb)[s] = p[(h8 + (s & 7) + ((s >> 3) << 4)) * 32];
        sfrag[j] = __builtin_amdgcn_wmma_f32_16x16x32_bf16(
            false, qf[dc].v, false, kb.v, (short)0, sfrag[j], false, false);
      }
    }

    // scale, causal mask, online softmax, write P to LDS
#pragma unroll
    for (int r = 0; r < 8; ++r) {
      const int mg = q0 + r + 8 * hi;
      float v0 = sfrag[0][r] * scale;
      float v1 = sfrag[1][r] * scale;
      if (kc0 + lm > mg)      v0 = -1.0e9f;
      if (kc0 + 16 + lm > mg) v1 = -1.0e9f;
      float bm = fmaxf(v0, v1);
      bm = fmaxf(bm, __shfl_xor(bm, 1));
      bm = fmaxf(bm, __shfl_xor(bm, 2));
      bm = fmaxf(bm, __shfl_xor(bm, 4));
      bm = fmaxf(bm, __shfl_xor(bm, 8));
      const float mnew  = fmaxf(mrow[r], bm);
      const float alpha = __expf(mrow[r] - mnew);
      mrow[r] = mnew;
      const float p0 = __expf(v0 - mnew);
      const float p1 = __expf(v1 - mnew);
      float rs = p0 + p1;
      rs += __shfl_xor(rs, 1);
      rs += __shfl_xor(rs, 2);
      rs += __shfl_xor(rs, 4);
      rs += __shfl_xor(rs, 8);
      lrow[r] = lrow[r] * alpha + rs;
#pragma unroll
      for (int j = 0; j < 8; ++j) o[j][r] = o[j][r] * alpha;
      ldsP[wave][(r + 8 * hi) * 32 + lm]      = f2bf(p0);
      ldsP[wave][(r + 8 * hi) * 32 + 16 + lm] = f2bf(p1);
    }

    // O += P V (wave reads only its own P area; same-wave LDS ordering)
    FragBF pf;
    {
      const unsigned short* p = &ldsP[wave][lm * 32 + h8];
      pf.q[0] = *(const uint4*)(p);
      pf.q[1] = *(const uint4*)(p + 16);
    }
#pragma unroll
    for (int j = 0; j < 8; ++j) {
      FragBF vb;
      const unsigned short* p = &ldsV[16 * j + lm];
#pragma unroll
      for (int s = 0; s < 16; ++s)
        ((unsigned short*)&vb)[s] = p[(h8 + (s & 7) + ((s >> 3) << 4)) * HD];
      o[j] = __builtin_amdgcn_wmma_f32_16x16x32_bf16(
          false, pf.v, false, vb.v, (short)0, o[j], false, false);
    }
    __syncthreads();
  }

  // normalize and store attn output as [B,S,h,d] bf16
#pragma unroll
  for (int j = 0; j < 8; ++j)
#pragma unroll
    for (int r = 0; r < 8; ++r) {
      const int m = r + 8 * hi;
      const int d = 16 * j + lm;
      const float val = o[j][r] / lrow[r];
      Om[((size_t)(b * S_DIM + q0 + m) * NH + h) * HD + d] = f2bf(val);
    }
}

// ---------------------------------------------------------------------------
// Host side
// ---------------------------------------------------------------------------
extern "C" void kernel_launch(void* const* d_in, const int* in_sizes, int n_in,
                              void* d_out, int out_size, void* d_ws,
                              size_t ws_size, hipStream_t stream) {
  (void)in_sizes; (void)n_in; (void)out_size; (void)ws_size;
  const float* hidden = (const float*)d_in[0];
  const float* cosb   = (const float*)d_in[2];
  const float* sinb   = (const float*)d_in[3];
  const float* wq     = (const float*)d_in[4];
  const float* wk     = (const float*)d_in[5];
  const float* wv     = (const float*)d_in[6];
  const float* wo     = (const float*)d_in[7];
  float* out = (float*)d_out;

  char* ws = (char*)d_ws;  // workspace layout (bytes)
  unsigned short* h_bf  = (unsigned short*)(ws + 0);            // 4096x4096
  unsigned short* wqT   = (unsigned short*)(ws + 33554432ull);  // [K][4096]
  unsigned short* wkT   = (unsigned short*)(ws + 67108864ull);  // [K][1024]
  unsigned short* wvT   = (unsigned short*)(ws + 75497472ull);  // [K][1024]
  unsigned short* woT   = (unsigned short*)(ws + 83886080ull);  // [K][4096]
  unsigned short* q_lin = (unsigned short*)(ws + 117440512ull); // 4096x4096
  unsigned short* k_lin = (unsigned short*)(ws + 150994944ull); // 4096x1024
  unsigned short* v_lin = (unsigned short*)(ws + 159383552ull); // 4096x1024
  unsigned short* q_r   = (unsigned short*)(ws + 167772160ull); // [B,32,S,128]
  unsigned short* k_r   = (unsigned short*)(ws + 201326592ull); // [B,8,S,128]
  unsigned short* v_r   = (unsigned short*)(ws + 209715200ull); // [B,8,S,128]
  unsigned short* attn  = (unsigned short*)(ws + 218103808ull); // 4096x4096

  // 0: one-time dtype/layout prep
  cvt_f32_bf16<<<(M_TOT * H_DIM / 8) / 256, 256, 0, stream>>>(hidden, h_bf);
  cvt_w_transpose<<<(4096 * 4096 / 4) / 256, 256, 0, stream>>>(wq, wqT, 4096,
                                                               4096);
  cvt_w_transpose<<<(1024 * 4096 / 4) / 256, 256, 0, stream>>>(wk, wkT, 1024,
                                                               4096);
  cvt_w_transpose<<<(1024 * 4096 / 4) / 256, 256, 0, stream>>>(wv, wvT, 1024,
                                                               4096);
  cvt_w_transpose<<<(4096 * 4096 / 4) / 256, 256, 0, stream>>>(wo, woT, 4096,
                                                               4096);

  // 1-3: QKV projections (TDM-staged bf16 WMMA GEMM, bf16 outputs)
  gemm_bf16_wmma<true><<<dim3(NH * HD / 128, M_TOT / 128), 256, 0, stream>>>(
      h_bf, wqT, q_lin, M_TOT, NH * HD, H_DIM);
  gemm_bf16_wmma<true><<<dim3(NKV * HD / 128, M_TOT / 128), 256, 0, stream>>>(
      h_bf, wkT, k_lin, M_TOT, NKV * HD, H_DIM);
  gemm_bf16_wmma<true><<<dim3(NKV * HD / 128, M_TOT / 128), 256, 0, stream>>>(
      h_bf, wvT, v_lin, M_TOT, NKV * HD, H_DIM);

  // 4: RoPE + transpose to [B,h,S,d]
  {
    const int tq = B_DIM * NH * S_DIM * 64;
    const int tk = B_DIM * NKV * S_DIM * 64;
    rope_transpose<<<tq / 256, 256, 0, stream>>>(q_lin, cosb, sinb, q_r, NH, 1);
    rope_transpose<<<tk / 256, 256, 0, stream>>>(k_lin, cosb, sinb, k_r, NKV, 1);
    rope_transpose<<<tk / 256, 256, 0, stream>>>(v_lin, cosb, sinb, v_r, NKV, 0);
  }

  // 5: causal flash attention
  flash_attn<<<dim3(S_DIM / 64, B_DIM * NH), 128, 0, stream>>>(q_r, k_r, v_r,
                                                               attn);

  // 6: output projection (fp32 result -> d_out)
  gemm_bf16_wmma<false><<<dim3(H_DIM / 128, M_TOT / 128), 256, 0, stream>>>(
      attn, woT, out, M_TOT, H_DIM, H_DIM);
}
